// Encoder_GAE_74887049773203
// MI455X (gfx1250) — compile-verified, matrix operand order checked by reading, not run on hardware
//
#include <hip/hip_runtime.h>
#include <stdint.h>

// ---------------------------------------------------------------------------
// GCN encoder for MI455X (gfx1250, wave32).
// GEMMs run on v_wmma_f32_16x16x32_bf16; edge aggregation uses native f32
// global atomics; self-loop/bias/relu/residual fused into one epilogue that
// also produces the bf16 operand for the next layer's WMMA GEMM.
// ---------------------------------------------------------------------------

typedef __attribute__((ext_vector_type(16))) __bf16 v16bf;
typedef __attribute__((ext_vector_type(8)))  float  v8f;

union Frag16 {
    v16bf    v;
    uint32_t u[8];
};

// ---------------------------------------------------------------------------
// Degree accumulation: deg[dst] += 1 per edge (self-loop added in finalize).
// ---------------------------------------------------------------------------
__global__ void deg_accum_kernel(const long long* __restrict__ dst,
                                 float* __restrict__ deg, int nEdges) {
    int tid = blockIdx.x * blockDim.x + threadIdx.x;
    if (tid >= nEdges) return;
    long long d = dst[tid];
    __hip_atomic_fetch_add(&deg[d], 1.0f, __ATOMIC_RELAXED,
                           __HIP_MEMORY_SCOPE_AGENT);
}

__global__ void deg_finalize_kernel(float* __restrict__ dinv, int nNodes) {
    int tid = blockIdx.x * blockDim.x + threadIdx.x;
    if (tid >= nNodes) return;
    float d = dinv[tid] + 1.0f;          // +1 self loop; always > 0
    dinv[tid] = rsqrtf(d);               // D^{-1/2}
}

// ---------------------------------------------------------------------------
// Weight convert: f32 [K x N] row-major  ->  bf16 transposed [N x K].
// Transposed layout makes every WMMA B-fragment VGPR one contiguous dword.
// ---------------------------------------------------------------------------
__global__ void weight_cvt_kernel(const float* __restrict__ W,
                                  __bf16* __restrict__ Bt, int K, int N) {
    int tid = blockIdx.x * blockDim.x + threadIdx.x;
    if (tid >= K * N) return;
    int k = tid / N;
    int n = tid - k * N;
    Bt[(size_t)n * K + k] = (__bf16)W[tid];
}

// f32 -> bf16 elementwise (initial x conversion)
__global__ void f32_to_bf16_kernel(const float* __restrict__ X,
                                   __bf16* __restrict__ Xb, int total) {
    int tid = blockIdx.x * blockDim.x + threadIdx.x;
    if (tid >= total) return;
    Xb[tid] = (__bf16)X[tid];
}

// ---------------------------------------------------------------------------
// bf16 WMMA GEMM:  D[M x N] = A[M x K] * B[K x N]   (B given transposed NxK)
// 256 threads = 8 waves; each wave computes a 16(M) x 64(N) strip via four
// 16x16 accumulators, K-loop in steps of 32 (v_wmma_f32_16x16x32_bf16).
// M % 16 == 0, K % 32 == 0, N % 64 == 0 guaranteed by the problem sizes.
// ---------------------------------------------------------------------------
__global__ __launch_bounds__(256)
void gemm_bf16_wmma_kernel(const __bf16* __restrict__ A,
                           const __bf16* __restrict__ Bt,
                           float* __restrict__ D,
                           int M, int K, int N) {
    const int lane  = threadIdx.x & 31;
    const int wave  = threadIdx.x >> 5;
    const int mTile = blockIdx.x * 8 + wave;
    if (mTile * 16 >= M) return;            // wave-uniform exit (EXEC stays all-1)
    const int mBase = mTile * 16;
    const int nBase = blockIdx.y * 64;

    const int mr = lane & 15;               // row (A) / col (B) within tile
    const int kh = (lane >> 4) * 8;         // lane-half K offset

    v8f acc0 = {}, acc1 = {}, acc2 = {}, acc3 = {};

    const __bf16* aRow = A  + (size_t)(mBase + mr) * K + kh;
    const __bf16* bR0  = Bt + (size_t)(nBase +  0 + mr) * K + kh;
    const __bf16* bR1  = Bt + (size_t)(nBase + 16 + mr) * K + kh;
    const __bf16* bR2  = Bt + (size_t)(nBase + 32 + mr) * K + kh;
    const __bf16* bR3  = Bt + (size_t)(nBase + 48 + mr) * K + kh;

    for (int k0 = 0; k0 < K; k0 += 32) {
        __builtin_prefetch(aRow + k0 + 32, 0, 1);   // global_prefetch_b8
        Frag16 fa, f0, f1, f2, f3;
#pragma unroll
        for (int v = 0; v < 8; ++v) {
            // VGPR v holds K = 16*(v>>2) + kh + 2*(v&3) + {0,1}
            const int ko = k0 + 16 * (v >> 2) + 2 * (v & 3);
            fa.u[v] = *(const uint32_t*)(aRow + ko);
            f0.u[v] = *(const uint32_t*)(bR0 + ko);
            f1.u[v] = *(const uint32_t*)(bR1 + ko);
            f2.u[v] = *(const uint32_t*)(bR2 + ko);
            f3.u[v] = *(const uint32_t*)(bR3 + ko);
        }
        acc0 = __builtin_amdgcn_wmma_f32_16x16x32_bf16(false, fa.v, false, f0.v,
                                                       (short)0, acc0, false, false);
        acc1 = __builtin_amdgcn_wmma_f32_16x16x32_bf16(false, fa.v, false, f1.v,
                                                       (short)0, acc1, false, false);
        acc2 = __builtin_amdgcn_wmma_f32_16x16x32_bf16(false, fa.v, false, f2.v,
                                                       (short)0, acc2, false, false);
        acc3 = __builtin_amdgcn_wmma_f32_16x16x32_bf16(false, fa.v, false, f3.v,
                                                       (short)0, acc3, false, false);
    }

    // C/D layout: lanes 0-15 rows M=r, lanes 16-31 rows M=r+8; N = lane&15.
    const int nc = lane & 15;
    const int mo = (lane >> 4) * 8;
#pragma unroll
    for (int r = 0; r < 8; ++r) {
        const size_t row = (size_t)(mBase + mo + r) * N + nBase;
        D[row +  0 + nc] = acc0[r];
        D[row + 16 + nc] = acc1[r];
        D[row + 32 + nc] = acc2[r];
        D[row + 48 + nc] = acc3[r];
    }
}

// ---------------------------------------------------------------------------
// Edge scatter: agg[dst] += dinv[src]*dinv[dst] * lin[src]
// One thread per (edge, 4 channels); channels fastest-varying -> each wave
// reads one coalesced float4 strip of lin[src] and issues row-local atomics.
// ---------------------------------------------------------------------------
__global__ __launch_bounds__(256)
void gcn_scatter_kernel(const long long* __restrict__ srcIdx,
                        const long long* __restrict__ dstIdx,
                        const float* __restrict__ lin,
                        const float* __restrict__ dinv,
                        float* __restrict__ agg,
                        int nEdges, int C) {
    const int chunks = C >> 2;
    unsigned tid = blockIdx.x * blockDim.x + threadIdx.x;
    unsigned total = (unsigned)nEdges * (unsigned)chunks;
    if (tid >= total) return;
    const int e  = tid / chunks;
    const int c4 = (tid - e * chunks) << 2;

    const long long s = srcIdx[e];
    const long long d = dstIdx[e];
    const float w = dinv[s] * dinv[d];

    const float4 hv = *(const float4*)(lin + (size_t)s * C + c4);
    float* ap = agg + (size_t)d * C + c4;
    __hip_atomic_fetch_add(ap + 0, w * hv.x, __ATOMIC_RELAXED, __HIP_MEMORY_SCOPE_AGENT);
    __hip_atomic_fetch_add(ap + 1, w * hv.y, __ATOMIC_RELAXED, __HIP_MEMORY_SCOPE_AGENT);
    __hip_atomic_fetch_add(ap + 2, w * hv.z, __ATOMIC_RELAXED, __HIP_MEMORY_SCOPE_AGENT);
    __hip_atomic_fetch_add(ap + 3, w * hv.w, __ATOMIC_RELAXED, __HIP_MEMORY_SCOPE_AGENT);
}

// ---------------------------------------------------------------------------
// Fused epilogue: out = [relu](agg + dinv^2 * lin + bias) [+ hprev]
// Optionally also emits bf16 copy (input operand of the next WMMA GEMM).
// ---------------------------------------------------------------------------
__global__ __launch_bounds__(256)
void gcn_epilogue_kernel(const float* __restrict__ agg,
                         const float* __restrict__ lin,
                         const float* __restrict__ hprev,
                         const float* __restrict__ bias,
                         const float* __restrict__ dinv,
                         float* __restrict__ hout,
                         __bf16* __restrict__ hbf,
                         int nNodes, int C, int doRelu, int doRes) {
    unsigned tid = blockIdx.x * blockDim.x + threadIdx.x;
    unsigned total = (unsigned)nNodes * (unsigned)C;
    if (tid >= total) return;
    const int i = tid / C;
    const int c = tid - i * C;
    const float di = dinv[i];
    float v = agg[tid] + di * di * lin[tid] + bias[c];
    if (doRelu) v = fmaxf(v, 0.0f);
    if (doRes)  v += hprev[tid];
    hout[tid] = v;
    if (hbf) hbf[tid] = (__bf16)v;
}

// ---------------------------------------------------------------------------
// Host-side orchestration
// ---------------------------------------------------------------------------
extern "C" void kernel_launch(void* const* d_in, const int* in_sizes, int n_in,
                              void* d_out, int out_size, void* d_ws, size_t ws_size,
                              hipStream_t stream) {
    const float*     x  = (const float*)d_in[0];
    const long long* ei = (const long long*)d_in[1];   // int64 per reference
    const float*     W1 = (const float*)d_in[2];
    const float*     b1 = (const float*)d_in[3];
    const float*     Wr = (const float*)d_in[4];
    const float*     br = (const float*)d_in[5];
    const float*     Wx = (const float*)d_in[6];
    const float*     bx = (const float*)d_in[7];
    float*           out = (float*)d_out;

    const int Cin = 512, H1c = 512, H2c = 256;
    const int Nn = in_sizes[0] / Cin;        // 50000
    const int nE = in_sizes[1] / 2;          // 800000
    const long long* srcIdx = ei;            // edge_index[0]
    const long long* dstIdx = ei + nE;       // edge_index[1]

    // --- carve workspace ---------------------------------------------------
    char* wp = (char*)d_ws;
    auto carve = [&](size_t bytes) -> void* {
        void* p = (void*)wp;
        wp += (bytes + 255) & ~(size_t)255;
        return p;
    };
    float*  dinv = (float*)carve((size_t)Nn * 4);
    float*  lin  = (float*)carve((size_t)Nn * H1c * 4);   // GEMM output (h = xW)
    float*  agg  = (float*)carve((size_t)Nn * H1c * 4);   // scatter accumulator
    float*  hbufF= (float*)carve((size_t)Nn * H1c * 4);   // current h (f32, residual)
    __bf16* hbufB= (__bf16*)carve((size_t)Nn * H1c * 2);  // current h (bf16 for WMMA)
    __bf16* w1t  = (__bf16*)carve((size_t)Cin * H1c * 2);
    __bf16* wrt  = (__bf16*)carve((size_t)H1c * H1c * 2);
    __bf16* wxt  = (__bf16*)carve((size_t)H1c * H2c * 2);

    // --- degree / normalization -------------------------------------------
    hipMemsetAsync(dinv, 0, (size_t)Nn * 4, stream);
    deg_accum_kernel<<<(nE + 255) / 256, 256, 0, stream>>>(dstIdx, dinv, nE);
    deg_finalize_kernel<<<(Nn + 255) / 256, 256, 0, stream>>>(dinv, Nn);

    // --- weights to transposed bf16 ---------------------------------------
    weight_cvt_kernel<<<(Cin * H1c + 255) / 256, 256, 0, stream>>>(W1, w1t, Cin, H1c);
    weight_cvt_kernel<<<(H1c * H1c + 255) / 256, 256, 0, stream>>>(Wr, wrt, H1c, H1c);
    weight_cvt_kernel<<<(H1c * H2c + 255) / 256, 256, 0, stream>>>(Wx, wxt, H1c, H2c);

    // --- x to bf16 (first GEMM operand) ------------------------------------
    f32_to_bf16_kernel<<<((Nn * Cin) + 255) / 256, 256, 0, stream>>>(x, hbufB, Nn * Cin);

    const int mTiles = Nn / 16;              // 3125 (exact)

    auto gcn_layer = [&](const __bf16* wT, const float* bias, int Cout,
                         int relu, int res, float* hOutF, __bf16* hOutB) {
        dim3 gg((mTiles + 7) / 8, Cout / 64);
        gemm_bf16_wmma_kernel<<<gg, 256, 0, stream>>>(hbufB, wT, lin, Nn, H1c, Cout);
        hipMemsetAsync(agg, 0, (size_t)Nn * Cout * 4, stream);
        unsigned scatTotal = (unsigned)nE * (unsigned)(Cout / 4);
        gcn_scatter_kernel<<<(scatTotal + 255) / 256, 256, 0, stream>>>(
            srcIdx, dstIdx, lin, dinv, agg, nE, Cout);
        unsigned epiTotal = (unsigned)Nn * (unsigned)Cout;
        gcn_epilogue_kernel<<<(epiTotal + 255) / 256, 256, 0, stream>>>(
            agg, lin, hbufF, bias, dinv, hOutF, hOutB, Nn, Cout, relu, res);
    };

    // Layer 1: h = relu(conv(x, W1, b1))
    gcn_layer(w1t, b1, H1c, /*relu=*/1, /*res=*/0, hbufF, hbufB);
    // 4 shared residual layers: h = relu(conv(h, Wr, br)) + h
    for (int l = 0; l < 4; ++l)
        gcn_layer(wrt, br, H1c, /*relu=*/1, /*res=*/1, hbufF, hbufB);
    // Output layer: out = conv(h, Wx, bx)  (no relu, no residual, f32 out)
    gcn_layer(wxt, bx, H2c, /*relu=*/0, /*res=*/0, out, nullptr);
}
